// ContourMerge_83399674954443
// MI455X (gfx1250) — matrix operand order repaired and checked by reading, not compiled, not run on hardware
//
#include <hip/hip_runtime.h>

// ---------------------------------------------------------------------------
// GCN (3 conv layers) + edge MLP for MI455X (gfx1250, wave32, WMMA bf16)
// Aggregation uses a CSR built by dst (no float atomics in the hot loops).
// ---------------------------------------------------------------------------

typedef unsigned short u16;
typedef unsigned int   u32;

typedef __attribute__((ext_vector_type(16))) __bf16 v16bf;
typedef __attribute__((ext_vector_type(8)))  __bf16 v8bf;
typedef __attribute__((ext_vector_type(8)))  float  v8f;

union AFrag { v16bf v; v8bf h[2]; };

__device__ __forceinline__ u16 f2bf(float f) {
    u32 u = __builtin_bit_cast(u32, f);
    u32 r = u + 0x7FFFu + ((u >> 16) & 1u);   // round-to-nearest-even
    return (u16)(r >> 16);
}

// ----------------------------- utility kernels -----------------------------

__global__ void fill_zero_u32(u32* __restrict__ p, long long n) {
    long long i = (long long)blockIdx.x * blockDim.x + threadIdx.x;
    if (i < n) p[i] = 0u;
}

__global__ void deg_count(const long long* __restrict__ dst, u32* __restrict__ cnt, int E) {
    int i = blockIdx.x * blockDim.x + threadIdx.x;
    if (i < E) atomicAdd(&cnt[(int)dst[i]], 1u);
}

__global__ void compute_dinv(const u32* __restrict__ cnt, float* __restrict__ dinv, int N) {
    int i = blockIdx.x * blockDim.x + threadIdx.x;
    if (i < N) dinv[i] = rsqrtf((float)cnt[i] + 1.0f);   // +1 = self loop
}

__global__ void cvt_f32_bf16(const float* __restrict__ src, u16* __restrict__ dst, long long n) {
    long long i = (long long)blockIdx.x * blockDim.x + threadIdx.x;
    if (i < n) dst[i] = f2bf(src[i]);
}

// --------------------- chunked exclusive prefix sum ------------------------
// N=100000 split into 1024 chunks of L; 3 trivial kernels (one-time cost).

#define NCHUNK 1024

__global__ void chunk_sum(const u32* __restrict__ cnt, u32* __restrict__ sums, int N, int L) {
    int c = blockIdx.x * blockDim.x + threadIdx.x;
    if (c >= NCHUNK) return;
    u32 s = 0;
    for (int j = 0; j < L; ++j) {
        int idx = c * L + j;
        if (idx < N) s += cnt[idx];
    }
    sums[c] = s;
}

__global__ void scan_sums(u32* __restrict__ sums, u32* __restrict__ rowptr, int N) {
    if (blockIdx.x == 0 && threadIdx.x == 0) {
        u32 run = 0;
        for (int c = 0; c < NCHUNK; ++c) { u32 t = sums[c]; sums[c] = run; run += t; }
        rowptr[N] = run;   // == E
    }
}

__global__ void chunk_scan(const u32* __restrict__ cnt, const u32* __restrict__ sums,
                           u32* __restrict__ rowptr, int N, int L) {
    int c = blockIdx.x * blockDim.x + threadIdx.x;
    if (c >= NCHUNK) return;
    u32 run = sums[c];
    for (int j = 0; j < L; ++j) {
        int idx = c * L + j;
        if (idx < N) { rowptr[idx] = run; run += cnt[idx]; }
    }
}

__global__ void copy_u32(const u32* __restrict__ src, u32* __restrict__ dst, int n) {
    int i = blockIdx.x * blockDim.x + threadIdx.x;
    if (i < n) dst[i] = src[i];
}

// Bin edges by dst; precompute per-edge norm = dinv[src]*dinv[dst]
// (reused unchanged by all three conv layers).
__global__ void bin_edges(const long long* __restrict__ src, const long long* __restrict__ dst,
                          const float* __restrict__ dinv, u32* __restrict__ pos,
                          int* __restrict__ csr_src, float* __restrict__ csr_w, int E) {
    int e = blockIdx.x * blockDim.x + threadIdx.x;
    if (e >= E) return;
    int s = (int)src[e], d = (int)dst[e];
    u32 slot = atomicAdd(&pos[d], 1u);
    csr_src[slot] = s;
    csr_w[slot]   = dinv[s] * dinv[d];
}

// ------------------------------- WMMA GEMM ---------------------------------
// C[N x COUT] (f32) = A[N x CIN] (bf16, row-major) @ Bpre (pre-swizzled bf16)
// One wave computes one 16x16 tile; 4 WMMAs per tile at CIN=128.

__global__ void pack_b(const float* __restrict__ W, u16* __restrict__ Bp,
                       int K, int Nsrc, int Npad) {
    int NT = Npad / 16, KB = K / 32;
    int total = KB * NT * 32 * 16;
    int i = blockIdx.x * blockDim.x + threadIdx.x;
    if (i >= total) return;
    int q    = i & 15;
    int lane = (i >> 4) & 31;
    int tn   = (i >> 9) % NT;
    int kb   = (i >> 9) / NT;
    int k = kb * 32 + (q >> 3) * 16 + (lane >> 4) * 8 + (q & 7);
    int n = tn * 16 + (lane & 15);
    float v = (n < Nsrc) ? W[(long long)k * Nsrc + n] : 0.0f;
    Bp[i] = f2bf(v);
}

template<int CIN, int COUT>
__launch_bounds__(256)
__global__ void gemm_bf16(const u16* __restrict__ Abits, const u16* __restrict__ Bbits,
                          float* __restrict__ Cmat, int Nn) {
    constexpr int NT = COUT / 16;
    constexpr int KB = CIN / 32;
    const __bf16* A  = reinterpret_cast<const __bf16*>(Abits);
    const v16bf*  Bp = reinterpret_cast<const v16bf*>(Bbits);

    int gid  = blockIdx.x * blockDim.x + threadIdx.x;
    int wid  = gid >> 5;
    int lane = gid & 31;
    int nw   = (gridDim.x * blockDim.x) >> 5;
    int tiles = (Nn / 16) * NT;   // N=100000 is a multiple of 16

    for (int t = wid; t < tiles; t += nw) {
        int tm = t / NT, tn = t % NT;
        int row = tm * 16 + (lane & 15);
        int klo = (lane >> 4) * 8;
        const __bf16* Ar = A + (long long)row * CIN;

        v8f c = {0.f, 0.f, 0.f, 0.f, 0.f, 0.f, 0.f, 0.f};
        #pragma unroll
        for (int kb = 0; kb < KB; ++kb) {
            AFrag af;
            af.h[0] = *reinterpret_cast<const v8bf*>(Ar + kb * 32 + klo);
            af.h[1] = *reinterpret_cast<const v8bf*>(Ar + kb * 32 + 16 + klo);
            v16bf b = Bp[(kb * NT + tn) * 32 + lane];
            c = __builtin_amdgcn_wmma_f32_16x16x32_bf16(false, af.v, false, b,
                                                        (short)0, c, false, false);
        }
        int n  = tn * 16 + (lane & 15);
        int mb = (lane >> 4) * 8;
        float* Cp = Cmat + (long long)(tm * 16) * COUT + n;
        #pragma unroll
        for (int r = 0; r < 8; ++r)
            Cp[(long long)(mb + r) * COUT] = c[r];
    }
}

// ----------------------- CSR gather aggregation ----------------------------
// out_i = sum_{j in in(i)} w_ij * h_j + dinv_i^2 * h_i ; then (+bias, ReLU?).
// Wave per node; lane covers VEC consecutive channels. CSR metadata is
// wave-uniform -> scalar loads. No atomics; fully load-streamed from L2.

__launch_bounds__(256)
__global__ void agg_relu_bf16(const float* __restrict__ h, const u32* __restrict__ rowptr,
                              const int* __restrict__ csr_src, const float* __restrict__ csr_w,
                              const float* __restrict__ dinv, const float* __restrict__ bias,
                              u16* __restrict__ out, int Nn) {
    constexpr int C = 128;
    int gid  = blockIdx.x * blockDim.x + threadIdx.x;
    int wid  = gid >> 5;
    int lane = gid & 31;
    int nw   = (gridDim.x * blockDim.x) >> 5;

    for (int i = wid; i < Nn; i += nw) {
        float di = dinv[i];
        float sw = di * di;                       // self-loop weight
        float4 v = *reinterpret_cast<const float4*>(h + (long long)i * C + lane * 4);
        float4 a;
        a.x = v.x * sw; a.y = v.y * sw; a.z = v.z * sw; a.w = v.w * sw;

        u32 k = rowptr[i], end = rowptr[i + 1];
        for (; k + 1 < end; k += 2) {             // 2-way unroll for ILP
            int   s0 = csr_src[k],     s1 = csr_src[k + 1];
            float w0 = csr_w[k],       w1 = csr_w[k + 1];
            float4 v0 = *reinterpret_cast<const float4*>(h + (long long)s0 * C + lane * 4);
            float4 v1 = *reinterpret_cast<const float4*>(h + (long long)s1 * C + lane * 4);
            a.x += v0.x * w0; a.y += v0.y * w0; a.z += v0.z * w0; a.w += v0.w * w0;
            a.x += v1.x * w1; a.y += v1.y * w1; a.z += v1.z * w1; a.w += v1.w * w1;
        }
        if (k < end) {
            int s = csr_src[k]; float w = csr_w[k];
            float4 v0 = *reinterpret_cast<const float4*>(h + (long long)s * C + lane * 4);
            a.x += v0.x * w; a.y += v0.y * w; a.z += v0.z * w; a.w += v0.w * w;
        }

        float4 bz = *reinterpret_cast<const float4*>(bias + lane * 4);
        ushort4 o;
        o.x = f2bf(fmaxf(a.x + bz.x, 0.0f));
        o.y = f2bf(fmaxf(a.y + bz.y, 0.0f));
        o.z = f2bf(fmaxf(a.z + bz.z, 0.0f));
        o.w = f2bf(fmaxf(a.w + bz.w, 0.0f));
        *reinterpret_cast<ushort4*>(out + (long long)i * C + lane * 4) = o;
    }
}

__launch_bounds__(256)
__global__ void agg_f32(const float* __restrict__ h, const u32* __restrict__ rowptr,
                        const int* __restrict__ csr_src, const float* __restrict__ csr_w,
                        const float* __restrict__ dinv, const float* __restrict__ bias,
                        float* __restrict__ out, int Nn) {
    constexpr int C = 64;
    int gid  = blockIdx.x * blockDim.x + threadIdx.x;
    int wid  = gid >> 5;
    int lane = gid & 31;
    int nw   = (gridDim.x * blockDim.x) >> 5;

    for (int i = wid; i < Nn; i += nw) {
        float di = dinv[i];
        float sw = di * di;
        float2 v = *reinterpret_cast<const float2*>(h + (long long)i * C + lane * 2);
        float2 a; a.x = v.x * sw; a.y = v.y * sw;

        u32 k = rowptr[i], end = rowptr[i + 1];
        for (; k + 1 < end; k += 2) {
            int   s0 = csr_src[k],     s1 = csr_src[k + 1];
            float w0 = csr_w[k],       w1 = csr_w[k + 1];
            float2 v0 = *reinterpret_cast<const float2*>(h + (long long)s0 * C + lane * 2);
            float2 v1 = *reinterpret_cast<const float2*>(h + (long long)s1 * C + lane * 2);
            a.x += v0.x * w0; a.y += v0.y * w0;
            a.x += v1.x * w1; a.y += v1.y * w1;
        }
        if (k < end) {
            int s = csr_src[k]; float w = csr_w[k];
            float2 v0 = *reinterpret_cast<const float2*>(h + (long long)s * C + lane * 2);
            a.x += v0.x * w; a.y += v0.y * w;
        }

        float2 bz = *reinterpret_cast<const float2*>(bias + lane * 2);
        float2 o; o.x = a.x + bz.x; o.y = a.y + bz.y;   // no ReLU (layer 2)
        *reinterpret_cast<float2*>(out + (long long)i * C + lane * 2) = o;
    }
}

// -------------------------------- edge MLP ---------------------------------
// Per wave: 16 edges. ef = (h2[src]+h2[dst])*0.5 -> LDS (bf16) -> WMMA 16x64 @
// 64x64 (+b1, ReLU) -> LDS -> WMMA 16x64 @ 64x16(padded w2) -> y[e, 0..1].
__launch_bounds__(256)
__global__ void edge_mlp(const float* __restrict__ h2,
                         const long long* __restrict__ src0,
                         const long long* __restrict__ dst0,
                         const u16* __restrict__ Bp1bits, const float* __restrict__ b1,
                         const u16* __restrict__ Bp2bits, const float* __restrict__ b2,
                         float* __restrict__ y, int E) {
    __shared__ __align__(16) u16 lds[8][16][72];   // 16 rows x 64 bf16 (+8 pad) per wave
    const v16bf* Bp1 = reinterpret_cast<const v16bf*>(Bp1bits);
    const v16bf* Bp2 = reinterpret_cast<const v16bf*>(Bp2bits);

    int lane = threadIdx.x & 31;
    int wv   = threadIdx.x >> 5;
    int nloc = lane & 15;
    int mb   = (lane >> 4) * 8;
    int klo  = (lane >> 4) * 8;
    int tiles = (E + 15) / 16;

    for (int bt = blockIdx.x * 8; bt < tiles; bt += gridDim.x * 8) {
        int t = bt + wv;
        bool act = t < tiles;

        // ---- gather 16 edge-feature rows into LDS as bf16 ----
        if (act) {
            int el = lane >> 1;             // local edge 0..15
            int hf = (lane & 1) * 32;       // feature half 0 / 32
            int e  = t * 16 + el;
            u16* row = &lds[wv][el][hf];
            if (e < E) {
                long long s = src0[e], d = dst0[e];
                const float4* ps = reinterpret_cast<const float4*>(h2 + (long long)s * 64 + hf);
                const float4* pd = reinterpret_cast<const float4*>(h2 + (long long)d * 64 + hf);
                #pragma unroll
                for (int j = 0; j < 8; ++j) {
                    float4 a = ps[j], b = pd[j];
                    row[j * 4 + 0] = f2bf((a.x + b.x) * 0.5f);
                    row[j * 4 + 1] = f2bf((a.y + b.y) * 0.5f);
                    row[j * 4 + 2] = f2bf((a.z + b.z) * 0.5f);
                    row[j * 4 + 3] = f2bf((a.w + b.w) * 0.5f);
                }
            } else {
                #pragma unroll
                for (int j = 0; j < 32; ++j) row[j] = 0;
            }
        }
        __syncthreads();

        // ---- layer 1: 16x64 @ 64x64, bias + ReLU, back to LDS as bf16 ----
        if (act) {
            v8f c[4];
            #pragma unroll
            for (int nt = 0; nt < 4; ++nt) c[nt] = (v8f){0.f,0.f,0.f,0.f,0.f,0.f,0.f,0.f};
            #pragma unroll
            for (int kb = 0; kb < 2; ++kb) {
                AFrag af;
                af.h[0] = *reinterpret_cast<const v8bf*>(&lds[wv][nloc][kb * 32 + klo]);
                af.h[1] = *reinterpret_cast<const v8bf*>(&lds[wv][nloc][kb * 32 + 16 + klo]);
                #pragma unroll
                for (int nt = 0; nt < 4; ++nt) {
                    v16bf b = Bp1[(kb * 4 + nt) * 32 + lane];
                    c[nt] = __builtin_amdgcn_wmma_f32_16x16x32_bf16(false, af.v, false, b,
                                                                    (short)0, c[nt], false, false);
                }
            }
            #pragma unroll
            for (int nt = 0; nt < 4; ++nt) {
                float bias = b1[nt * 16 + nloc];
                #pragma unroll
                for (int r = 0; r < 8; ++r) {
                    float v = fmaxf(c[nt][r] + bias, 0.0f);
                    lds[wv][mb + r][nt * 16 + nloc] = f2bf(v);
                }
            }
        }
        __syncthreads();

        // ---- layer 2: 16x64 @ 64x16(padded), store columns 0..1 ----
        if (act) {
            v8f c2 = {0.f,0.f,0.f,0.f,0.f,0.f,0.f,0.f};
            #pragma unroll
            for (int kb = 0; kb < 2; ++kb) {
                AFrag af;
                af.h[0] = *reinterpret_cast<const v8bf*>(&lds[wv][nloc][kb * 32 + klo]);
                af.h[1] = *reinterpret_cast<const v8bf*>(&lds[wv][nloc][kb * 32 + 16 + klo]);
                v16bf b = Bp2[kb * 32 + lane];
                c2 = __builtin_amdgcn_wmma_f32_16x16x32_bf16(false, af.v, false, b,
                                                             (short)0, c2, false, false);
            }
            if (nloc < 2) {
                float bias = b2[nloc];
                #pragma unroll
                for (int r = 0; r < 8; ++r) {
                    int e = t * 16 + mb + r;
                    if (e < E) y[(long long)e * 2 + nloc] = c2[r] + bias;
                }
            }
        }
        __syncthreads();
    }
}

// ------------------------------ host launcher ------------------------------

extern "C" void kernel_launch(void* const* d_in, const int* in_sizes, int n_in,
                              void* d_out, int out_size, void* d_ws, size_t ws_size,
                              hipStream_t stream) {
    const float*     x    = (const float*)d_in[0];
    const long long* ei   = (const long long*)d_in[1];
    const float*     W0   = (const float*)d_in[2];
    const float*     b0   = (const float*)d_in[3];
    const float*     W1   = (const float*)d_in[4];
    const float*     b1   = (const float*)d_in[5];
    const float*     W2   = (const float*)d_in[6];
    const float*     b2   = (const float*)d_in[7];
    const float*     mw1  = (const float*)d_in[8];
    const float*     mb1  = (const float*)d_in[9];
    const float*     mw2  = (const float*)d_in[10];
    const float*     mb2  = (const float*)d_in[11];
    float*           y    = (float*)d_out;

    const int N = in_sizes[0] / 128;     // 100000 (multiple of 16)
    const int E = in_sizes[1] / 2;       // 1600000
    const long long* src0 = ei;
    const long long* dst0 = ei + E;

    char* ws = (char*)d_ws;
    size_t off = 0;
    auto take = [&](size_t bytes) -> char* {
        char* p = ws + off;
        off += (bytes + 255) & ~(size_t)255;
        return p;
    };
    u32*   cnt     = (u32*)  take((size_t)N * 4);
    float* dinv    = (float*)take((size_t)N * 4);
    u32*   pos     = (u32*)  take((size_t)N * 4);
    u32*   rowptr  = (u32*)  take((size_t)(N + 1) * 4);
    u32*   sums    = (u32*)  take((size_t)NCHUNK * 4);
    int*   csr_src = (int*)  take((size_t)E * 4);
    float* csr_w   = (float*)take((size_t)E * 4);
    u16*   a_bf    = (u16*)  take((size_t)N * 128 * 2);
    float* hbuf    = (float*)take((size_t)N * 128 * 4);   // reused as N x 64 for layer 2 + MLP
    float* accb    = (float*)take((size_t)N * 128 * 4);   // layer-2 GEMM output (ping-pong with hbuf)
    u16*   Bp0     = (u16*)  take(128 * 128 * 2);
    u16*   Bp1     = (u16*)  take(128 * 128 * 2);
    u16*   Bp2     = (u16*)  take(128 * 64 * 2);
    u16*   BpM1    = (u16*)  take(64 * 64 * 2);
    u16*   BpM2    = (u16*)  take(64 * 16 * 2);

    const int T = 256;
    auto blks = [](long long n, int t) { return (int)((n + t - 1) / t); };
    const int L = (N + NCHUNK - 1) / NCHUNK;

    // ---- degrees -> dinv, CSR build (shared by all 3 conv layers) ----
    fill_zero_u32<<<blks(N, T), T, 0, stream>>>(cnt, N);
    deg_count<<<blks(E, T), T, 0, stream>>>(dst0, cnt, E);
    compute_dinv<<<blks(N, T), T, 0, stream>>>(cnt, dinv, N);

    chunk_sum<<<NCHUNK / T, T, 0, stream>>>(cnt, sums, N, L);
    scan_sums<<<1, 1, 0, stream>>>(sums, rowptr, N);
    chunk_scan<<<NCHUNK / T, T, 0, stream>>>(cnt, sums, rowptr, N, L);
    copy_u32<<<blks(N, T), T, 0, stream>>>(rowptr, pos, N);
    bin_edges<<<blks(E, T), T, 0, stream>>>(src0, dst0, dinv, pos, csr_src, csr_w, E);

    // ---- weight pre-swizzle (bf16 WMMA B-fragments) ----
    pack_b<<<blks(128 / 32 * 128 / 16 * 512, T), T, 0, stream>>>(W0,  Bp0, 128, 128, 128);
    pack_b<<<blks(128 / 32 * 128 / 16 * 512, T), T, 0, stream>>>(W1,  Bp1, 128, 128, 128);
    pack_b<<<blks(128 / 32 * 64  / 16 * 512, T), T, 0, stream>>>(W2,  Bp2, 128, 64,  64);
    pack_b<<<blks(64  / 32 * 64  / 16 * 512, T), T, 0, stream>>>(mw1, BpM1, 64, 64,  64);
    pack_b<<<blks(64  / 32 * 16  / 16 * 512, T), T, 0, stream>>>(mw2, BpM2, 64, 2,   16);

    // ---- x -> bf16 activations ----
    cvt_f32_bf16<<<blks((long long)N * 128, T), T, 0, stream>>>(x, a_bf, (long long)N * 128);

    const int tiles128  = (N / 16) * (128 / 16);
    const int tiles64   = (N / 16) * (64 / 16);
    const int AGG_BLOCKS = blks((long long)N * 32, T);   // wave per node

    // ---- layer 0: GEMM -> CSR aggregate (+b0, ReLU, ->bf16) ----
    gemm_bf16<128, 128><<<(tiles128 + 7) / 8, T, 0, stream>>>(a_bf, Bp0, hbuf, N);
    agg_relu_bf16<<<AGG_BLOCKS, T, 0, stream>>>(hbuf, rowptr, csr_src, csr_w, dinv, b0, a_bf, N);

    // ---- layer 1 ----
    gemm_bf16<128, 128><<<(tiles128 + 7) / 8, T, 0, stream>>>(a_bf, Bp1, hbuf, N);
    agg_relu_bf16<<<AGG_BLOCKS, T, 0, stream>>>(hbuf, rowptr, csr_src, csr_w, dinv, b1, a_bf, N);

    // ---- layer 2 (no ReLU, f32 output for MLP) ----
    gemm_bf16<128, 64><<<(tiles64 + 7) / 8, T, 0, stream>>>(a_bf, Bp2, accb, N);
    agg_f32<<<AGG_BLOCKS, T, 0, stream>>>(accb, rowptr, csr_src, csr_w, dinv, b2, hbuf, N);

    // ---- edge MLP -> y ----
    int etiles = (E + 15) / 16;
    edge_mlp<<<(etiles + 7) / 8, T, 0, stream>>>(hbuf, src0, dst0, BpM1, mb1, BpM2, mb2, y, E);
}
